// MeshMeanFlowNet_51316269252796
// MI455X (gfx1250) — compile-verified, matrix operand order checked by reading, not run on hardware
//
#include <hip/hip_runtime.h>
#include <math.h>

// ---------- types ----------
typedef __attribute__((ext_vector_type(16))) __bf16 v16bf;
typedef __attribute__((ext_vector_type(8)))  float  v8f;
typedef unsigned int uint32x4 __attribute__((ext_vector_type(4)));
typedef int          int32x8  __attribute__((ext_vector_type(8)));
typedef int          int32x4  __attribute__((ext_vector_type(4)));

union ABu {
    v16bf v;
    unsigned short s[16];
    unsigned int   u[8];
};

__device__ __forceinline__ unsigned short f2bf(float f) {
    unsigned int u = __float_as_uint(f);
    unsigned int r = u + 0x7FFFu + ((u >> 16) & 1u);   // RNE
    return (unsigned short)(r >> 16);
}
__device__ __forceinline__ float bf2f(unsigned short s) {
    return __uint_as_float(((unsigned int)s) << 16);
}

__device__ __forceinline__ v8f wmma_bf16(const ABu& a, const ABu& b, v8f c) {
    return __builtin_amdgcn_wmma_f32_16x16x32_bf16(
        /*neg_a=*/false, a.v, /*neg_b=*/false, b.v,
        /*c_mod=*/(short)0, c, /*reuse_a=*/false, /*reuse_b=*/false);
}

// TDM: DMA a 64(rows) x 64(k) bf16 tile of a row-major (N x K) tensor into LDS.
// LDS destination is padded: 16 bytes of pad per 128-byte row (pad_interval=32dw, pad_amount=4dw)
// so the LDS row stride is 72 ushorts.
__device__ __forceinline__ void tdm_load_tile64x64(
    const unsigned short* gsrc, unsigned lds_off, int K, int N) {
    unsigned long long ga = (unsigned long long)(size_t)gsrc;
    uint32x4 g0;
    g0[0] = 1u;                                   // count=1, user mode
    g0[1] = lds_off;                              // lds_addr
    g0[2] = (unsigned)ga;                         // global_addr[31:0]
    g0[3] = (unsigned)(ga >> 32) | (2u << 30);    // global_addr[56:32] | type=2
    int32x8 g1;
    g1[0] = (int)((1u << 16)                      // data_size = 2 bytes
                  | (1u << 20)                    // pad_enable
                  | (4u << 22)                    // pad_interval: 32 DWORDs
                  | (3u << 25));                  // pad_amount: 4 DWORDs (16B)
    g1[1] = (int)((unsigned)(K & 0xFFFF) << 16);                      // tensor_dim0 lo
    g1[2] = (int)(((unsigned)(K >> 16) & 0xFFFF) |
                  ((unsigned)(N & 0xFFFF) << 16));                    // dim0 hi | dim1 lo
    g1[3] = (int)(((unsigned)(N >> 16) & 0xFFFF) | (64u << 16));      // dim1 hi | tile_dim0=64
    g1[4] = 64;                                   // tile_dim1=64, tile_dim2=0
    g1[5] = K;                                    // tensor_dim0_stride lo32 (elements)
    g1[6] = 0;                                    // stride hi | dim1_stride lo
    g1[7] = 0;
    int32x4 z4 = {0, 0, 0, 0};
    int32x8 z8 = {0, 0, 0, 0, 0, 0, 0, 0};
    __builtin_amdgcn_tensor_load_to_lds(g0, g1, z4, z4, z8, 0);
}

// ---------- tiny kernels ----------

// p[b, 0:1024] = silu(cond[b,:]) @ w (512x1024) + bias
__global__ __launch_bounds__(256) void adaparam_kernel(
    const float* __restrict__ cond, const float* __restrict__ w,
    const float* __restrict__ bias, float* __restrict__ p) {
    __shared__ float sc[512];
    int b = blockIdx.x, tid = threadIdx.x;
    for (int k = tid; k < 512; k += 256) {
        float c = cond[b * 512 + k];
        sc[k] = c / (1.0f + __expf(-c));
    }
    __syncthreads();
    for (int cg = 0; cg < 4; ++cg) {
        int col = tid + 256 * cg;
        float acc = bias[col];
        for (int k = 0; k < 512; ++k) acc = fmaf(sc[k], w[k * 1024 + col], acc);
        p[b * 1024 + col] = acc;
    }
}

// fp32 KxN -> bf16 NxK (transposed)
__global__ __launch_bounds__(256) void cvt_t_kernel(
    const float* __restrict__ src, unsigned short* __restrict__ dst, int K, int N) {
    int idx = blockIdx.x * 256 + threadIdx.x;
    if (idx >= K * N) return;
    int k = idx / N, n = idx % N;
    dst[(size_t)n * K + k] = f2bf(src[idx]);
}

// per-row layernorm (D=512) with adaLN modulation -> bf16
__global__ __launch_bounds__(256) void ln_mod_kernel(
    const float* __restrict__ x, const float* __restrict__ p,
    unsigned short* __restrict__ h, int V) {
    __shared__ float red[256];
    int row = blockIdx.x, tid = threadIdx.x;
    int b = row / V;
    const float* xr = x + (size_t)row * 512;
    float v0 = xr[tid], v1 = xr[tid + 256];
    red[tid] = v0 + v1;
    __syncthreads();
    for (int s = 128; s > 0; s >>= 1) {
        if (tid < s) red[tid] += red[tid + s];
        __syncthreads();
    }
    float mean = red[0] * (1.0f / 512.0f);
    __syncthreads();
    float d0 = v0 - mean, d1 = v1 - mean;
    red[tid] = d0 * d0 + d1 * d1;
    __syncthreads();
    for (int s = 128; s > 0; s >>= 1) {
        if (tid < s) red[tid] += red[tid + s];
        __syncthreads();
    }
    float rs = rsqrtf(red[0] * (1.0f / 512.0f) + 1e-5f);
    const float* pb = p + (size_t)b * 1024;
    int c0 = tid, c1 = tid + 256;
    float y0 = d0 * rs * (1.0f + pb[c0]) + pb[512 + c0];
    float y1 = d1 * rs * (1.0f + pb[c1]) + pb[512 + c1];
    h[(size_t)row * 512 + c0] = f2bf(y0);
    h[(size_t)row * 512 + c1] = f2bf(y1);
}

// ---------- WMMA GEMM with TDM-staged weights ----------
// C = A(bf16 MxK) @ Wt(bf16 NxK pre-transposed).  One block (4 waves) per 64x64 C tile.
// Weight panel (64 rows x 64 k) is DMA'd into LDS by the Tensor Data Mover, double buffered.
// mode 0: store bf16;  mode 1: +bias, exact GELU, store bf16;  mode 2: +bias +res, store f32
__global__ __launch_bounds__(128) void gemm_wmma_tdm_kernel(
    const unsigned short* __restrict__ A, const unsigned short* __restrict__ Wt,
    const float* __restrict__ bias, const float* __restrict__ res,
    void* __restrict__ out, int M, int N, int K, int mode) {
    __shared__ __align__(16) unsigned short Bt[2][64][72];  // 16B pad per 128B row

    int wave = threadIdx.x >> 5, lane = threadIdx.x & 31;
    int nGroups = N >> 6;
    int t = blockIdx.x;
    int mt = t / nGroups, ng = t % nGroups;
    int r = lane & 15, ksel = lane >> 4;

    const unsigned short* wpanel = Wt + (size_t)(ng * 64) * K;
    const unsigned short* arow = A + (size_t)(mt * 64 + wave * 16 + r) * K;

    v8f acc[4];
    acc[0] = {}; acc[1] = {}; acc[2] = {}; acc[3] = {};

    int nChunks = K >> 6;
    // prologue: DMA first weight panel chunk into buffer 0
    if (wave == 0) {
        tdm_load_tile64x64(wpanel, (unsigned)(size_t)&Bt[0][0][0], K, N);
    }
    for (int c = 0; c < nChunks; ++c) {
        int cur = c & 1;
        int kc = c << 6;
        __builtin_amdgcn_s_wait_tensorcnt(0);   // panel[cur] resident (issuing wave)
        __syncthreads();                        // publish to all waves; prior reads of panel[1-cur] done
        if (wave == 0 && (c + 1) < nChunks) {
            tdm_load_tile64x64(wpanel + (size_t)(kc + 64),
                               (unsigned)(size_t)&Bt[1 - cur][0][0], K, N);
        }
        // A operands for this 64-K chunk (two 16x32 sub-chunks), direct global b128 loads
        ABu a0, a1, bm;
        ((uint4*)a0.u)[0] = *(const uint4*)(arow + kc + 8 * ksel);
        ((uint4*)a0.u)[1] = *(const uint4*)(arow + kc + 16 + 8 * ksel);
        ((uint4*)a1.u)[0] = *(const uint4*)(arow + kc + 32 + 8 * ksel);
        ((uint4*)a1.u)[1] = *(const uint4*)(arow + kc + 48 + 8 * ksel);
#pragma unroll
        for (int bt = 0; bt < 4; ++bt) {
            const unsigned short* brow = &Bt[cur][bt * 16 + r][0];
            ((uint4*)bm.u)[0] = *(const uint4*)(brow + 8 * ksel);
            ((uint4*)bm.u)[1] = *(const uint4*)(brow + 16 + 8 * ksel);
            acc[bt] = wmma_bf16(a0, bm, acc[bt]);
            ((uint4*)bm.u)[0] = *(const uint4*)(brow + 32 + 8 * ksel);
            ((uint4*)bm.u)[1] = *(const uint4*)(brow + 48 + 8 * ksel);
            acc[bt] = wmma_bf16(a1, bm, acc[bt]);
        }
    }
#pragma unroll
    for (int bt = 0; bt < 4; ++bt) {
#pragma unroll
        for (int j = 0; j < 8; ++j) {
            int m = mt * 64 + wave * 16 + j + 8 * ksel;
            int col = ng * 64 + bt * 16 + r;
            float v = acc[bt][j];
            if (mode == 0) {
                ((unsigned short*)out)[(size_t)m * N + col] = f2bf(v);
            } else if (mode == 1) {
                v += bias[col];
                v = 0.5f * v * (1.0f + erff(v * 0.70710678118654752f));
                ((unsigned short*)out)[(size_t)m * N + col] = f2bf(v);
            } else {
                v += bias[col] + res[(size_t)m * N + col];
                ((float*)out)[(size_t)m * N + col] = v;
            }
        }
    }
}

// ---------- attention: one WG (4 waves) per (b, h, 16-query tile) ----------
__global__ __launch_bounds__(128) void attn_kernel(
    const unsigned short* __restrict__ qkv,  // (B,V,1536) bf16: [q|k|v] x (H=8, hd=64)
    const int* __restrict__ eidx,            // (B,V,V)
    const float* __restrict__ etab,          // (4,H)
    unsigned short* __restrict__ O,          // (B,V,512) bf16
    int V) {
    const int D3 = 1536, H = 8;
    __shared__ __align__(16) unsigned short S[16][1040];  // bf16 scores/probs, padded
    __shared__ float redmax[16][8];
    __shared__ float redsum[16][8];

    int qt = blockIdx.x, h = blockIdx.y, b = blockIdx.z;
    int wave = threadIdx.x >> 5, lane = threadIdx.x & 31;
    int r = lane & 15, ksel = lane >> 4;
    int q0 = qt * 16;

    // ---- load Q tile as two A-operand chunks (head dim 64 = 2 x K32) ----
    ABu qa0, qa1;
    {
        const unsigned short* qrow = qkv + ((size_t)(b * V + q0 + r)) * D3 + h * 64;
        ((uint4*)qa0.u)[0] = *(const uint4*)(qrow + 0 + 8 * ksel);
        ((uint4*)qa0.u)[1] = *(const uint4*)(qrow + 16 + 8 * ksel);
        ((uint4*)qa1.u)[0] = *(const uint4*)(qrow + 32 + 8 * ksel);
        ((uint4*)qa1.u)[1] = *(const uint4*)(qrow + 48 + 8 * ksel);
    }

    // ---- S = Q K^T : waves split the 64 key tiles ----
    for (int kt = wave; kt < (V >> 4); kt += 4) {
        int key0 = kt * 16;
        const unsigned short* krow =
            qkv + ((size_t)(b * V + key0 + r)) * D3 + 512 + h * 64;
        ABu kb;
        v8f c = {};
        ((uint4*)kb.u)[0] = *(const uint4*)(krow + 0 + 8 * ksel);
        ((uint4*)kb.u)[1] = *(const uint4*)(krow + 16 + 8 * ksel);
        c = wmma_bf16(qa0, kb, c);
        ((uint4*)kb.u)[0] = *(const uint4*)(krow + 32 + 8 * ksel);
        ((uint4*)kb.u)[1] = *(const uint4*)(krow + 48 + 8 * ksel);
        c = wmma_bf16(qa1, kb, c);
#pragma unroll
        for (int j = 0; j < 8; ++j)
            S[j + 8 * ksel][key0 + r] = f2bf(c[j]);
    }
    __syncthreads();

    // ---- softmax with edge bias: 8 threads per query row ----
    {
        int row = threadIdx.x & 15, g = threadIdx.x >> 4;
        const float scale = 0.125f;  // hd^-0.5
        const int* erow = eidx + ((size_t)b * V + (q0 + row)) * V;
        float bt0 = etab[0 * H + h], bt1 = etab[1 * H + h];
        float bt2 = etab[2 * H + h], bt3 = etab[3 * H + h];
        float mx = -3.0e38f;
        for (int col = g; col < V; col += 8) {
            int e = erow[col] & 3;
            float bia = (e == 0) ? bt0 : (e == 1) ? bt1 : (e == 2) ? bt2 : bt3;
            float v = bf2f(S[row][col]) * scale + bia;
            S[row][col] = f2bf(v);
            mx = fmaxf(mx, v);
        }
        redmax[row][g] = mx;
        __syncthreads();
#pragma unroll
        for (int i = 0; i < 8; ++i) mx = fmaxf(mx, redmax[row][i]);
        float sum = 0.0f;
        for (int col = g; col < V; col += 8) {
            float e = __expf(bf2f(S[row][col]) - mx);
            S[row][col] = f2bf(e);
            sum += e;
        }
        redsum[row][g] = sum;
        __syncthreads();
        sum = 0.0f;
#pragma unroll
        for (int i = 0; i < 8; ++i) sum += redsum[row][i];
        float inv = 1.0f / sum;
        for (int col = g; col < V; col += 8)
            S[row][col] = f2bf(bf2f(S[row][col]) * inv);
    }
    __syncthreads();

    // ---- O = P @ V : each wave owns 16 of the 64 head-dim columns ----
    int ng = wave;
    v8f o = {};
    ABu pa, vb;
    for (int kc = 0; kc < V; kc += 32) {
        // P rows are contiguous bf16 in LDS -> two ds_load_b128 per lane
        const unsigned short* prow = &S[r][kc];
        ((uint4*)pa.u)[0] = *(const uint4*)(prow + 8 * ksel);
        ((uint4*)pa.u)[1] = *(const uint4*)(prow + 16 + 8 * ksel);
#pragma unroll
        for (int e = 0; e < 16; ++e) {
            int kk = (e < 8) ? (8 * ksel + e) : (16 + 8 * ksel + (e - 8));
            vb.s[e] = qkv[((size_t)(b * V + kc + kk)) * D3 + 1024 + h * 64 + ng * 16 + r];
        }
        o = wmma_bf16(pa, vb, o);
    }
#pragma unroll
    for (int j = 0; j < 8; ++j) {
        int m = j + 8 * ksel;
        O[((size_t)(b * V) + q0 + m) * 512 + h * 64 + ng * 16 + r] = f2bf(o[j]);
    }
}

// ---------- host ----------
extern "C" void kernel_launch(void* const* d_in, const int* in_sizes, int n_in,
                              void* d_out, int out_size, void* d_ws, size_t ws_size,
                              hipStream_t stream) {
    const int B = 8, V = 1024, H = 8;
    const int BV = B * V;  // 8192 rows

    const float* x      = (const float*)d_in[0];
    const float* cond   = (const float*)d_in[1];
    const int*   eidx   = (const int*)d_in[2];
    const float* w_ada1 = (const float*)d_in[3];
    const float* b_ada1 = (const float*)d_in[4];
    const float* w_ada2 = (const float*)d_in[5];
    const float* b_ada2 = (const float*)d_in[6];
    const float* w_qkv  = (const float*)d_in[7];
    const float* w_proj = (const float*)d_in[8];
    const float* b_proj = (const float*)d_in[9];
    const float* etab   = (const float*)d_in[10];
    const float* w_mlp1 = (const float*)d_in[11];
    const float* b_mlp1 = (const float*)d_in[12];
    const float* w_mlp2 = (const float*)d_in[13];
    const float* b_mlp2 = (const float*)d_in[14];

    char* ws = (char*)d_ws;
    auto alloc = [&](size_t bytes) -> void* {
        void* p = (void*)ws;
        ws += (bytes + 255) & ~(size_t)255;
        return p;
    };
    float*          p1      = (float*)alloc((size_t)B * 1024 * 4);
    float*          p2      = (float*)alloc((size_t)B * 1024 * 4);
    unsigned short* wqkv_t  = (unsigned short*)alloc((size_t)1536 * 512 * 2);
    unsigned short* wproj_t = (unsigned short*)alloc((size_t)512 * 512 * 2);
    unsigned short* wmlp1_t = (unsigned short*)alloc((size_t)2048 * 512 * 2);
    unsigned short* wmlp2_t = (unsigned short*)alloc((size_t)512 * 2048 * 2);
    unsigned short* h1      = (unsigned short*)alloc((size_t)BV * 512 * 2);
    unsigned short* qkv     = (unsigned short*)alloc((size_t)BV * 1536 * 2);
    unsigned short* Obuf    = (unsigned short*)alloc((size_t)BV * 512 * 2);
    float*          x2      = (float*)alloc((size_t)BV * 512 * 4);
    unsigned short* h2      = (unsigned short*)alloc((size_t)BV * 512 * 2);
    unsigned short* m1      = (unsigned short*)alloc((size_t)BV * 2048 * 2);

    // adaLN parameter vectors
    adaparam_kernel<<<B, 256, 0, stream>>>(cond, w_ada1, b_ada1, p1);
    adaparam_kernel<<<B, 256, 0, stream>>>(cond, w_ada2, b_ada2, p2);

    // weight convert + transpose to bf16 NxK
    cvt_t_kernel<<<(512 * 1536 + 255) / 256, 256, 0, stream>>>(w_qkv, wqkv_t, 512, 1536);
    cvt_t_kernel<<<(512 * 512 + 255) / 256, 256, 0, stream>>>(w_proj, wproj_t, 512, 512);
    cvt_t_kernel<<<(512 * 2048 + 255) / 256, 256, 0, stream>>>(w_mlp1, wmlp1_t, 512, 2048);
    cvt_t_kernel<<<(2048 * 512 + 255) / 256, 256, 0, stream>>>(w_mlp2, wmlp2_t, 2048, 512);

    // h1 = adaln1(x)
    ln_mod_kernel<<<BV, 256, 0, stream>>>(x, p1, h1, V);

    // qkv = h1 @ w_qkv   (M=8192, N=1536, K=512)
    gemm_wmma_tdm_kernel<<<(BV / 64) * (1536 / 64), 128, 0, stream>>>(
        h1, wqkv_t, nullptr, nullptr, (void*)qkv, BV, 1536, 512, 0);

    // attention
    attn_kernel<<<dim3(V / 16, H, B), 128, 0, stream>>>(qkv, eidx, etab, Obuf, V);

    // x2 = x + O @ w_proj + b_proj   (M=8192, N=512, K=512)
    gemm_wmma_tdm_kernel<<<(BV / 64) * (512 / 64), 128, 0, stream>>>(
        Obuf, wproj_t, b_proj, x, (void*)x2, BV, 512, 512, 2);

    // h2 = adaln2(x2)
    ln_mod_kernel<<<BV, 256, 0, stream>>>(x2, p2, h2, V);

    // m1 = gelu(h2 @ w_mlp1 + b_mlp1)   (M=8192, N=2048, K=512)
    gemm_wmma_tdm_kernel<<<(BV / 64) * (2048 / 64), 128, 0, stream>>>(
        h2, wmlp1_t, b_mlp1, nullptr, (void*)m1, BV, 2048, 512, 1);

    // out = x2 + m1 @ w_mlp2 + b_mlp2   (M=8192, N=512, K=2048)
    gemm_wmma_tdm_kernel<<<(BV / 64) * (512 / 64), 128, 0, stream>>>(
        m1, wmlp2_t, b_mlp2, x2, d_out, BV, 512, 2048, 2);
}
